// TernaryConv2d_27522150433244
// MI455X (gfx1250) — matrix operand order repaired and checked by reading, not compiled, not run on hardware
//
#include <hip/hip_runtime.h>

// ---------------------------------------------------------------------------
// Types
// ---------------------------------------------------------------------------
typedef __attribute__((ext_vector_type(16))) __bf16  v16bf;
typedef __attribute__((ext_vector_type(8)))  float   v8f;
typedef __attribute__((ext_vector_type(4)))  float   vf4;
typedef __attribute__((ext_vector_type(8)))  unsigned short vu16x8;
typedef __attribute__((ext_vector_type(4)))  unsigned short vu16x4;

#define NELEM_W   (9 * 256 * 256)   // 589824 weight elements
#define WQ_STRIDE 2304              // 9*256 packed-K stride per output channel
#define H_IMG     56
#define W_IMG     56

// LDS slab: 3 halo rows x 18 halo cols = 54 positions, 256 ch, bf16 hi+lo.
// Row stride padded to 264 bf16 (528 B = 132 dwords, ≡4 mod 64 banks).
#define POS_STRIDE 264
#define N_POS      54

__device__ __forceinline__ unsigned short f32_to_bf16(float f) {
    unsigned int u = __float_as_uint(f);
    unsigned int r = u + 0x7FFFu + ((u >> 16) & 1u);   // round-to-nearest-even
    return (unsigned short)(r >> 16);
}
__device__ __forceinline__ float bf16_to_f32(unsigned short h) {
    return __uint_as_float(((unsigned int)h) << 16);
}

// ---------------------------------------------------------------------------
// ws layout: wsf[0]=sum|W|  wsf[1]=sum(a)  wsf[2]=count  wsf[3]=alpha
//            byte 256.. : packed ternary bf16 weights [co][tap*256+ci]
// ---------------------------------------------------------------------------

__global__ void zero_ws_kernel(float* wsf) {
    if (threadIdx.x < 4) wsf[threadIdx.x] = 0.0f;
}

__global__ void reduce_absW_kernel(const float* __restrict__ W, float* wsf) {
    __shared__ float sm[256];
    float s = 0.0f;
    for (int i = blockIdx.x * blockDim.x + threadIdx.x; i < NELEM_W;
         i += gridDim.x * blockDim.x)
        s += fabsf(W[i]);
    sm[threadIdx.x] = s;
    __syncthreads();
    for (int o = 128; o > 0; o >>= 1) {
        if ((int)threadIdx.x < o) sm[threadIdx.x] += sm[threadIdx.x + o];
        __syncthreads();
    }
    if (threadIdx.x == 0) atomicAdd(&wsf[0], sm[0]);
}

__global__ void reduce_alpha_kernel(const float* __restrict__ W, float* wsf) {
    __shared__ float sa[256];
    __shared__ float sn[256];
    const float t = 0.7f * wsf[0] / (float)NELEM_W;
    float suma = 0.0f, cnt = 0.0f;
    for (int i = blockIdx.x * blockDim.x + threadIdx.x; i < NELEM_W;
         i += gridDim.x * blockDim.x) {
        float w  = W[i];
        float aw = fabsf(w);
        if ((w > t) || (w < -t)) { suma += aw; cnt += 1.0f; }
    }
    sa[threadIdx.x] = suma;
    sn[threadIdx.x] = cnt;
    __syncthreads();
    for (int o = 128; o > 0; o >>= 1) {
        if ((int)threadIdx.x < o) {
            sa[threadIdx.x] += sa[threadIdx.x + o];
            sn[threadIdx.x] += sn[threadIdx.x + o];
        }
        __syncthreads();
    }
    if (threadIdx.x == 0) {
        atomicAdd(&wsf[1], sa[0]);
        atomicAdd(&wsf[2], sn[0]);
    }
}

// W is HWIO: idx = ((tap)*256 + ci)*256 + co.  Pack ternary bf16 as
// wqp[co*2304 + tap*256 + ci] so a lane's B fragment is two runs of 8 contiguous K values.
__global__ void quantize_pack_kernel(const float* __restrict__ W,
                                     const float* __restrict__ wsf_in,
                                     unsigned short* __restrict__ wqp,
                                     float* __restrict__ wsf_out) {
    const int idx = blockIdx.x * blockDim.x + threadIdx.x;
    const float t = 0.7f * wsf_in[0] / (float)NELEM_W;
    if (idx == 0) wsf_out[3] = wsf_in[1] / wsf_in[2];   // alpha
    if (idx < NELEM_W) {
        float w    = W[idx];
        int   co   = idx & 255;
        int   rest = idx >> 8;
        int   ci   = rest & 255;
        int   tap  = rest >> 8;
        // sign(sign(W+t)+sign(W-t)):  W>=t -> +1 ;  W<=-t -> -1 ; else 0
        unsigned short q = (w >= t) ? 0x3F80u : ((w <= -t) ? 0xBF80u : 0u);
        wqp[(size_t)co * WQ_STRIDE + tap * 256 + ci] = q;
    }
}

// ---------------------------------------------------------------------------
// Implicit-GEMM conv.  Block = 128 threads = 4 waves.
// Block tile: 16 output pixels (along W, fixed n,h) x 256 output channels.
// Wave w handles N-cols [64w, 64w+64) as four 16x16 f32 accum tiles.
// Stage ONCE per block: 3x18 halo positions x 256 ch, fp32 -> bf16 hi/lo in
// LDS.  Then 9 taps x 8 k-blocks: 4 ds_load_b128 (A hi/lo) + 8 b128 B loads
// + 8 v_wmma_f32_16x16x32_bf16 — no conversion VALU in the hot loop.
// ---------------------------------------------------------------------------
__global__ __launch_bounds__(128)
void conv_ternary_wmma_kernel(const float* __restrict__ x,
                              const unsigned short* __restrict__ wq,
                              const float* __restrict__ bias,
                              const float* __restrict__ wsf,
                              float* __restrict__ out) {
    __shared__ __align__(16) unsigned short lhs_hi[N_POS * POS_STRIDE];
    __shared__ __align__(16) unsigned short lhs_lo[N_POS * POS_STRIDE];

    const int tid  = threadIdx.x;
    const int blk  = blockIdx.x;
    const int wt   = blk & 3;                 // w-tile (0..3)
    const int h    = (blk >> 2) % H_IMG;
    const int n    = blk / (4 * H_IMG);
    const int w0   = wt * 16;
    const int lane = tid & 31;
    const int wave = tid >> 5;
    const int n0   = wave * 64;
    const int mrow = lane & 15;
    const int hiG  = (lane >> 4) & 1;         // lane group 16..31

    // ---- stage + convert halo slab once: 54 pos x 64 float4 = 3456 slots ---
    #pragma unroll
    for (int j = 0; j < 27; ++j) {
        int idx  = tid + j * 128;             // 0..3455
        int pos  = idx >> 6;                  // 0..53
        int c4   = idx & 63;                  // float4 channel group
        int row  = pos / 18;                  // halo row 0..2
        int wloc = pos - row * 18;            // halo col 0..17
        int hh   = h + row - 1;
        int ww   = w0 + wloc - 1;
        vf4 v = (vf4)0.0f;
        if ((unsigned)hh < (unsigned)H_IMG && (unsigned)ww < (unsigned)W_IMG) {
            size_t off = ((size_t)((n * H_IMG + hh) * W_IMG + ww) << 8) + c4 * 4;
            v = *(const vf4*)(x + off);
        }
        vu16x4 vhi, vlo;
        #pragma unroll
        for (int i = 0; i < 4; ++i) {
            unsigned short hb = f32_to_bf16(v[i]);
            vhi[i] = hb;
            vlo[i] = f32_to_bf16(v[i] - bf16_to_f32(hb));
        }
        *(vu16x4*)&lhs_hi[pos * POS_STRIDE + c4 * 4] = vhi;
        *(vu16x4*)&lhs_lo[pos * POS_STRIDE + c4 * 4] = vlo;
    }
    __syncthreads();

    v8f acc[4] = {};

    for (int tap = 0; tap < 9; ++tap) {
        const int dy = tap / 3 - 1;
        const int dx = tap % 3 - 1;
        // Lane's A row: halo position for output pixel m = mrow.
        const int posBase = ((dy + 1) * 18 + (mrow + dx + 1)) * POS_STRIDE;

        const unsigned short* wtap = wq + tap * 256;
        for (int kb = 0; kb < 8; ++kb) {
            // A fragment: lanes 0-15 take K {0..7,16..23}, lanes 16-31 take
            // K {8..15,24..31} of this k-block (16-bit A VGPR layout).
            const int abase = posBase + kb * 32 + (hiG ? 8 : 0);
            union { v16bf v; vu16x8 u[2]; } Ahi, Alo;
            Ahi.u[0] = *(const vu16x8*)&lhs_hi[abase];
            Ahi.u[1] = *(const vu16x8*)&lhs_hi[abase + 16];
            Alo.u[0] = *(const vu16x8*)&lhs_lo[abase];
            Alo.u[1] = *(const vu16x8*)&lhs_lo[abase + 16];

            const int ks = kb * 32 + (hiG ? 8 : 0);
            #pragma unroll
            for (int t = 0; t < 4; ++t) {
                const int nn = n0 + t * 16 + mrow;     // B: N = lane%16
                const unsigned short* bp =
                    wtap + (size_t)nn * WQ_STRIDE + ks;
                union { v16bf v; vu16x8 h[2]; } B;
                B.h[0] = *(const vu16x8*)(bp);
                B.h[1] = *(const vu16x8*)(bp + 16);
                acc[t] = __builtin_amdgcn_wmma_f32_16x16x32_bf16(
                    false, Ahi.v, false, B.v, (short)0, acc[t], false, false);
                acc[t] = __builtin_amdgcn_wmma_f32_16x16x32_bf16(
                    false, Alo.v, false, B.v, (short)0, acc[t], false, false);
            }
        }
    }

    // --- epilogue: y = acc*alpha + b, masked for w >= 56 -----------------
    const float alpha = wsf[3];
    #pragma unroll
    for (int t = 0; t < 4; ++t) {
        const int   col = n0 + t * 16 + mrow;
        const float bv  = bias[col];
        #pragma unroll
        for (int r = 0; r < 8; ++r) {
            int m = r + (hiG ? 8 : 0);        // C/D layout: lanes 16-31 => M+8
            int w = w0 + m;
            if (w < W_IMG) {
                size_t off = ((size_t)((n * H_IMG + h) * W_IMG + w) << 8) + col;
                out[off] = acc[t][r] * alpha + bv;
            }
        }
    }
}

// ---------------------------------------------------------------------------
extern "C" void kernel_launch(void* const* d_in, const int* in_sizes, int n_in,
                              void* d_out, int out_size, void* d_ws, size_t ws_size,
                              hipStream_t stream) {
    const float* x = (const float*)d_in[0];
    const float* W = (const float*)d_in[1];
    const float* b = (const float*)d_in[2];
    float*       y = (float*)d_out;

    float*          wsf = (float*)d_ws;
    unsigned short* wqp = (unsigned short*)((char*)d_ws + 256);

    (void)in_sizes; (void)n_in; (void)out_size; (void)ws_size;

    zero_ws_kernel<<<1, 32, 0, stream>>>(wsf);
    reduce_absW_kernel<<<288, 256, 0, stream>>>(W, wsf);
    reduce_alpha_kernel<<<288, 256, 0, stream>>>(W, wsf);
    quantize_pack_kernel<<<(NELEM_W + 255) / 256, 256, 0, stream>>>(W, wsf, wqp, wsf);

    // 32 batches * 56 rows * 4 w-tiles = 7168 blocks of 128 threads
    conv_ternary_wmma_kernel<<<32 * H_IMG * 4, 128, 0, stream>>>(x, wqp, b, wsf, y);
}
